// ExplainableGNN_34531537060047
// MI455X (gfx1250) — compile-verified, matrix operand order checked by reading, not run on hardware
//
#include <hip/hip_runtime.h>
#include <hip/hip_bf16.h>
#include <math.h>

typedef __attribute__((ext_vector_type(16))) _Float16 v16h;
typedef __attribute__((ext_vector_type(8)))  _Float16 v8h;
typedef __attribute__((ext_vector_type(8)))  float    v8f;

#define N_NODES 100000

// ---------------------------------------------------------------------------
// Generic f32 -> f16 conversion, 4 elements/thread (b128 load, b64 store).
// ---------------------------------------------------------------------------
__global__ void cvt_f32_to_f16(const float* __restrict__ in,
                               _Float16* __restrict__ out, long long n)
{
    const long long n4 = n >> 2;
    for (long long i = blockIdx.x * (long long)blockDim.x + threadIdx.x;
         i < n4; i += (long long)gridDim.x * blockDim.x) {
        const float4 v = ((const float4*)in)[i];
        v8h dummy; (void)dummy;
        _Float16 o0 = (_Float16)v.x, o1 = (_Float16)v.y;
        _Float16 o2 = (_Float16)v.z, o3 = (_Float16)v.w;
        typedef __attribute__((ext_vector_type(4))) _Float16 v4h;
        v4h o = {o0, o1, o2, o3};
        ((v4h*)out)[i] = o;
    }
    // tail (all our sizes are multiples of 4; kept for safety)
    if (blockIdx.x == 0 && threadIdx.x == 0) {
        for (long long i = n4 << 2; i < n; ++i) out[i] = (_Float16)in[i];
    }
}

// ---------------------------------------------------------------------------
// Dual-output WMMA GEMM:
//   Yl[n_rows,out_c] = Xh @ Wl^T ,  Yr[n_rows,out_c] = Xh @ Wr^T
// All matrices f16 row-major (Wl/Wr are [out_c, in_c]); f32 accumulate.
// One wave = one 16-row tile. Lane layout per CDNA5 ISA 7.12.2:
//   A (16x32): lanes 0-15 -> M=lane,  K = {k0..k0+7, k0+16..k0+23}
//              lanes 16-31 -> M=lane-16, K shifted by +8
//   B (32x16): same pattern with N in place of M (B = W^T -> read W rows)
//   D (16x16 f32): VGPR r: lanes 0-15 M=r, lanes 16-31 M=8+r; N = lane&15
// 16-byte alignment of every v8h load holds: row strides (128/64 halves) and
// K offsets (multiples of 8 halves) are multiples of 16 bytes.
// ---------------------------------------------------------------------------
__global__ void dual_gemm_wmma(const _Float16* __restrict__ Xh,
                               const _Float16* __restrict__ Wl,
                               const _Float16* __restrict__ Wr,
                               float* __restrict__ Yl,
                               float* __restrict__ Yr,
                               int n_rows, int in_c, int out_c)
{
    const int wave = blockIdx.x * (blockDim.x >> 5) + (threadIdx.x >> 5);
    const int m0   = wave * 16;
    if (m0 >= n_rows) return;              // wave-uniform: EXEC stays all-1s

    const int lane  = threadIdx.x & 31;
    const int mr    = lane & 15;
    const int khalf = (lane >> 4) * 8;

    const _Float16* xrow = Xh + (size_t)(m0 + mr) * in_c + khalf;

    for (int n0 = 0; n0 < out_c; n0 += 16) {
        v8f accL = {};
        v8f accR = {};
        const _Float16* wlrow = Wl + (size_t)(n0 + mr) * in_c + khalf;
        const _Float16* wrrow = Wr + (size_t)(n0 + mr) * in_c + khalf;

        for (int k0 = 0; k0 < in_c; k0 += 32) {
            const v8h a0 = *(const v8h*)(xrow + k0);
            const v8h a1 = *(const v8h*)(xrow + k0 + 16);
            const v8h bl0 = *(const v8h*)(wlrow + k0);
            const v8h bl1 = *(const v8h*)(wlrow + k0 + 16);
            const v8h br0 = *(const v8h*)(wrrow + k0);
            const v8h br1 = *(const v8h*)(wrrow + k0 + 16);
            const v16h a  = __builtin_shufflevector(a0, a1,
                0,1,2,3,4,5,6,7,8,9,10,11,12,13,14,15);
            const v16h bl = __builtin_shufflevector(bl0, bl1,
                0,1,2,3,4,5,6,7,8,9,10,11,12,13,14,15);
            const v16h br = __builtin_shufflevector(br0, br1,
                0,1,2,3,4,5,6,7,8,9,10,11,12,13,14,15);
            accL = __builtin_amdgcn_wmma_f32_16x16x32_f16(
                       false, a, false, bl, (short)0, accL, false, false);
            accR = __builtin_amdgcn_wmma_f32_16x16x32_f16(
                       false, a, false, br, (short)0, accR, false, false);
        }

        const int col   = n0 + mr;
        const int rbase = m0 + ((lane >> 4) * 8);
#pragma unroll
        for (int r = 0; r < 8; ++r) {
            Yl[(size_t)(rbase + r) * out_c + col] = accL[r];
            Yr[(size_t)(rbase + r) * out_c + col] = accR[r];
        }
    }
}

// ---------------------------------------------------------------------------
// Per-edge degree accumulation: deg[dst] += 1
// ---------------------------------------------------------------------------
__global__ void edge_degree(const long long* __restrict__ dst,
                            float* __restrict__ deg, long long n_edges)
{
    for (long long e = blockIdx.x * (long long)blockDim.x + threadIdx.x;
         e < n_edges; e += (long long)gridDim.x * blockDim.x) {
        unsafeAtomicAdd(&deg[(int)dst[e]], 1.0f);
    }
}

// ---------------------------------------------------------------------------
// Scatter-add: agg[dst, c..c+3] += Yin[src, c..c+3]
// One thread per (edge, 4-channel group): one b128 gather + 4 f32 atomics.
// Yin (<=25.6MB) and agg both L2-resident (192MB) -> atomics at L2 rate.
// ---------------------------------------------------------------------------
__global__ void edge_scatter_add4(const long long* __restrict__ src,
                                  const long long* __restrict__ dst,
                                  const float* __restrict__ Yin,
                                  float* __restrict__ agg,
                                  long long n_edges, int cshift)
{
    const int qshift = cshift - 2;                 // groups of 4 channels
    const long long total = n_edges << qshift;
    const int qmask = (1 << qshift) - 1;
    for (long long idx = blockIdx.x * (long long)blockDim.x + threadIdx.x;
         idx < total; idx += (long long)gridDim.x * blockDim.x) {
        const long long e = idx >> qshift;
        const int c = (int)(idx & qmask) << 2;
        const int s = (int)src[e];
        const int d = (int)dst[e];
        const float4 v = *(const float4*)(Yin + ((size_t)s << cshift) + c);
        float* base = agg + ((size_t)d << cshift) + c;
        unsafeAtomicAdd(base + 0, v.x);
        unsafeAtomicAdd(base + 1, v.y);
        unsafeAtomicAdd(base + 2, v.z);
        unsafeAtomicAdd(base + 3, v.w);
    }
}

// ---------------------------------------------------------------------------
// outh = (f16) relu(agg/max(deg,1) + bias + resid)   (f16: next GEMM's input)
// ---------------------------------------------------------------------------
__global__ void finalize_relu_f16(const float* __restrict__ agg,
                                  const float* __restrict__ resid,
                                  const float* __restrict__ bias,
                                  const float* __restrict__ deg,
                                  _Float16* __restrict__ outh,
                                  int n_nodes, int cshift)
{
    const long long total = (long long)n_nodes << cshift;
    const int cmask = (1 << cshift) - 1;
    for (long long idx = blockIdx.x * (long long)blockDim.x + threadIdx.x;
         idx < total; idx += (long long)gridDim.x * blockDim.x) {
        const int n = (int)(idx >> cshift);
        const int c = (int)(idx & cmask);
        const float dgn = fmaxf(deg[n], 1.0f);
        const float v = agg[idx] / dgn + bias[c] + resid[idx];
        outh[idx] = (_Float16)fmaxf(v, 0.0f);
    }
}

// ---------------------------------------------------------------------------
// logits = H(f16) @ W_lin^T + b_lin (2 outputs); out = log_softmax(logits)
// ---------------------------------------------------------------------------
__global__ void head_logsoftmax(const _Float16* __restrict__ H,  // [N, h2]
                                const float* __restrict__ Wlin,  // [2, h2]
                                const float* __restrict__ blin,  // [2]
                                float* __restrict__ out,         // [N, 2]
                                int n_nodes, int h2)
{
    for (int n = blockIdx.x * blockDim.x + threadIdx.x;
         n < n_nodes; n += gridDim.x * blockDim.x) {
        const _Float16* h = H + (size_t)n * h2;
        float l0 = blin[0], l1 = blin[1];
        for (int c = 0; c < h2; ++c) {
            const float hv = (float)h[c];
            l0 += hv * Wlin[c];
            l1 += hv * Wlin[h2 + c];
        }
        const float m   = fmaxf(l0, l1);
        const float lse = m + logf(expf(l0 - m) + expf(l1 - m));
        out[(size_t)n * 2 + 0] = l0 - lse;
        out[(size_t)n * 2 + 1] = l1 - lse;
    }
}

// ---------------------------------------------------------------------------
extern "C" void kernel_launch(void* const* d_in, const int* in_sizes, int n_in,
                              void* d_out, int out_size, void* d_ws, size_t ws_size,
                              hipStream_t stream)
{
    const float*     x    = (const float*)d_in[0];
    const long long* ei   = (const long long*)d_in[1];   // [2, E] int64
    const float*     W1l  = (const float*)d_in[2];
    const float*     b1l  = (const float*)d_in[3];
    const float*     W1r  = (const float*)d_in[4];
    const float*     W2l  = (const float*)d_in[5];
    const float*     b2l  = (const float*)d_in[6];
    const float*     W2r  = (const float*)d_in[7];
    const float*     Wlin = (const float*)d_in[8];
    const float*     blin = (const float*)d_in[9];
    float*           out  = (float*)d_out;

    const int N    = N_NODES;
    const int in_c = in_sizes[0] / N;            // 128
    const int hid  = in_sizes[2] / in_c;         // 64
    const int h2   = in_sizes[5] / hid;          // 32
    const long long E = in_sizes[1] / 2;         // 3.2M
    const long long* src = ei;
    const long long* dst = ei + E;
    const int csh1 = __builtin_ctz(hid);         // 6
    const int csh2 = __builtin_ctz(h2);          // 5

    // ---- workspace carve-up (256B aligned) ----
    char* ws = (char*)d_ws;
    size_t off = 0;
    auto carve = [&](size_t bytes) {
        void* p = (void*)(ws + off);
        off += (bytes + 255) & ~(size_t)255;
        return p;
    };
    _Float16* xh   = (_Float16*)carve((size_t)N * in_c * 2);   // x in f16
    _Float16* W1lh = (_Float16*)carve((size_t)hid * in_c * 2);
    _Float16* W1rh = (_Float16*)carve((size_t)hid * in_c * 2);
    _Float16* W2lh = (_Float16*)carve((size_t)h2 * hid * 2);
    _Float16* W2rh = (_Float16*)carve((size_t)h2 * hid * 2);
    float*    y1   = (float*)carve((size_t)N * hid * 4);       // x @ W1_l^T
    float*    r1   = (float*)carve((size_t)N * hid * 4);       // x @ W1_r^T
    float*    agg1 = (float*)carve((size_t)N * hid * 4);
    float*    deg  = (float*)carve((size_t)N * 4);
    _Float16* h1h  = (_Float16*)carve((size_t)N * hid * 2);    // relu out, f16
    float*    y2   = (float*)carve((size_t)N * h2 * 4);
    float*    r2   = (float*)carve((size_t)N * h2 * 4);
    float*    agg2 = (float*)carve((size_t)N * h2 * 4);
    _Float16* h2h  = (_Float16*)carve((size_t)N * h2 * 2);
    (void)ws_size;

    // ---- zero the accumulators (graph-capturable memsets) ----
    hipMemsetAsync(agg1, 0, (size_t)N * hid * sizeof(float), stream);
    hipMemsetAsync(deg,  0, (size_t)N * sizeof(float), stream);
    hipMemsetAsync(agg2, 0, (size_t)N * h2 * sizeof(float), stream);

    const int BLK = 256;
    const int wavesPerBlk = BLK / 32;
    const int rowTiles = (N + 15) / 16;                       // 6250
    const int gemmGrid = (rowTiles + wavesPerBlk - 1) / wavesPerBlk;

    // ---- one-time f16 conversions ----
    cvt_f32_to_f16<<<2048, BLK, 0, stream>>>(x,   xh,   (long long)N * in_c);
    cvt_f32_to_f16<<<8,    BLK, 0, stream>>>(W1l, W1lh, (long long)hid * in_c);
    cvt_f32_to_f16<<<8,    BLK, 0, stream>>>(W1r, W1rh, (long long)hid * in_c);
    cvt_f32_to_f16<<<2,    BLK, 0, stream>>>(W2l, W2lh, (long long)h2 * hid);
    cvt_f32_to_f16<<<2,    BLK, 0, stream>>>(W2r, W2rh, (long long)h2 * hid);

    // ---- layer 1: fused dual projection (WMMA) ----
    dual_gemm_wmma<<<gemmGrid, BLK, 0, stream>>>(xh, W1lh, W1rh, y1, r1,
                                                 N, in_c, hid);

    // ---- layer 1: edge aggregation ----
    edge_degree<<<2048, BLK, 0, stream>>>(dst, deg, E);
    edge_scatter_add4<<<8192, BLK, 0, stream>>>(src, dst, y1, agg1, E, csh1);

    // ---- layer 1: mean + bias + residual + relu -> h1 (f16) ----
    {
        long long tot = (long long)N << csh1;
        int grid = (int)((tot + BLK - 1) / BLK);
        finalize_relu_f16<<<grid, BLK, 0, stream>>>(agg1, r1, b1l, deg, h1h,
                                                    N, csh1);
    }

    // ---- layer 2: fused dual projection (WMMA) ----
    dual_gemm_wmma<<<gemmGrid, BLK, 0, stream>>>(h1h, W2lh, W2rh, y2, r2,
                                                 N, hid, h2);

    // ---- layer 2: edge aggregation ----
    edge_scatter_add4<<<8192, BLK, 0, stream>>>(src, dst, y2, agg2, E, csh2);

    // ---- layer 2: finalize -> h2 (f16) ----
    {
        long long tot = (long long)N << csh2;
        int grid = (int)((tot + BLK - 1) / BLK);
        finalize_relu_f16<<<grid, BLK, 0, stream>>>(agg2, r2, b2l, deg, h2h,
                                                    N, csh2);
    }

    // ---- head + log_softmax ----
    head_logsoftmax<<<(N + BLK - 1) / BLK, BLK, 0, stream>>>(
        h2h, Wlin, blin, out, N, h2);
}
